// TransformerBlock_71708773974456
// MI455X (gfx1250) — matrix operand order repaired and checked by reading, not compiled, not run on hardware
//
#include <hip/hip_runtime.h>

#define NT    4096
#define ED    1024
#define NH    16
#define HD    64
#define FFN   4096
#define NSEQ  8

typedef __bf16 bf16;
typedef __attribute__((ext_vector_type(8)))  __bf16 v8bf;
typedef __attribute__((ext_vector_type(16))) __bf16 v16bf;
typedef __attribute__((ext_vector_type(8)))  float  v8f;
typedef int v4i_vs __attribute__((vector_size(16)));   // matches builtin param type

union FragBF { v16bf v; v8bf h[2]; };

// ---- optional CDNA5 async global->LDS path (guarded; falls back to sync) ----
#if defined(__has_builtin)
#if __has_builtin(__builtin_amdgcn_global_load_async_to_lds_b128) && \
    __has_builtin(__builtin_amdgcn_s_wait_asynccnt)
#define HAVE_ASYNC_LDS 1
#endif
#endif

#if defined(HAVE_ASYNC_LDS)
__device__ __forceinline__ void async_cp16(const void* g, void* l) {
  __builtin_amdgcn_global_load_async_to_lds_b128((v4i_vs*)g, (v4i_vs*)l, 0, 0);
}
__device__ __forceinline__ void wait_async_all() { __builtin_amdgcn_s_wait_asynccnt(0); }
#else
__device__ __forceinline__ void wait_async_all() {}
#endif

__device__ __forceinline__ v8f wmma_bf16(v16bf a, v16bf b, v8f c) {
  return __builtin_amdgcn_wmma_f32_16x16x32_bf16(false, a, false, b, (short)0, c, false, false);
}

// A-fragment (16x32, MxK) from [m][k] LDS: lane<16 -> row=lane, kbase=0; lane>=16 -> kbase=8.
// elements 0..7 = K kbase..kbase+7 ; elements 8..15 = K kbase+16..kbase+23
__device__ __forceinline__ v16bf ldfragA(const bf16* base, int row, int stride, int k0, int lane) {
  const bf16* p = base + row * stride + k0 + ((lane < 16) ? 0 : 8);
  FragBF f;
  f.h[0] = *(const v8bf*)p;
  f.h[1] = *(const v8bf*)(p + 16);
  return f.v;
}

// B-fragment (32x16, KxN) from LDS stored transposed [n][k]:
// lane<16 -> col=lane, kbase=0; lane>=16 -> col=lane-16, kbase=8; elements 0..7 = K kbase..kbase+7, 8..15 = K kbase+16..kbase+23
__device__ __forceinline__ v16bf ldfragB(const bf16* baseT, int col, int stride, int k0, int lane) {
  const bf16* p = baseT + col * stride + k0 + ((lane < 16) ? 0 : 8);
  FragBF f;
  f.h[0] = *(const v8bf*)p;
  f.h[1] = *(const v8bf*)(p + 16);
  return f.v;
}

__device__ __forceinline__ float gelu_f(float x) {
  float x3 = x * x * x;
  return 0.5f * x * (1.0f + tanhf(0.7978845608028654f * (x + 0.044715f * x3)));
}

// ---------------------------------------------------------------- conversions
__global__ __launch_bounds__(256) void f32_to_bf16_k(const float* __restrict__ x,
                                                     bf16* __restrict__ y, int n) {
  int i = blockIdx.x * 256 + threadIdx.x;
  int stride = gridDim.x * 256;
  for (; i < n; i += stride) y[i] = (bf16)x[i];
}

// transpose+convert: X f32 [K][N] -> Y bf16 [N][K]
__global__ __launch_bounds__(256) void convT_k(const float* __restrict__ X,
                                               bf16* __restrict__ Y, int K, int N) {
  __shared__ float tile[32][33];
  int kb = blockIdx.y * 32, nb = blockIdx.x * 32;
  int tx = threadIdx.x & 31, ty = threadIdx.x >> 5;   // ty 0..7
#pragma unroll
  for (int i = ty; i < 32; i += 8) tile[i][tx] = X[(size_t)(kb + i) * N + nb + tx];
  __syncthreads();
#pragma unroll
  for (int i = ty; i < 32; i += 8) Y[(size_t)(nb + i) * K + kb + tx] = (bf16)tile[tx][i];
}

__global__ __launch_bounds__(256) void seg_k(const int* __restrict__ cu, int* __restrict__ seg) {
  int n = blockIdx.x * 256 + threadIdx.x;
  if (n >= NT) return;
  int s = 0;
#pragma unroll
  for (int i = 1; i <= NSEQ; ++i) s += (cu[i] <= n) ? 1 : 0;
  seg[n] = s;
}

// ------------------------------------------------------------------- GEMM
// C[M,N] = A[M,K](bf16) * B (pre-transposed Bt[N][K], bf16) + bias[N]
// Block tile 128(M) x 64(N), K-step 64, double-buffered LDS; 8 waves, 2x2 WMMA per wave.
// EP=0 -> f32 out, EP=1 -> bf16 gelu out
template <int EP>
__global__ __launch_bounds__(256) void gemm_bf16_k(const bf16* __restrict__ A,
                                                   const bf16* __restrict__ Bt,
                                                   const float* __restrict__ bias,
                                                   void* __restrict__ Cout,
                                                   int M, int N, int K) {
  __shared__ __align__(16) bf16 As[2][128 * 72];   // [m][k] stride 72
  __shared__ __align__(16) bf16 Bs[2][64 * 72];    // [n][k] stride 72
  int tid = threadIdx.x;
  int lane = tid & 31, wave = tid >> 5;
  int m0 = blockIdx.y * 128, n0 = blockIdx.x * 64;
  int wm = (wave >> 1) * 32;       // wave m-offset (0,32,64,96)
  int wn = (wave & 1) * 32;        // wave n-offset (0,32)

  v8f acc[2][2];
#pragma unroll
  for (int mi = 0; mi < 2; ++mi)
#pragma unroll
    for (int ni = 0; ni < 2; ++ni)
      acc[mi][ni] = (v8f){0.f, 0.f, 0.f, 0.f, 0.f, 0.f, 0.f, 0.f};

  int arow = tid >> 1, acg = (tid & 1) * 32;    // A loader: 128 rows x 64 halfs
  int bn = tid >> 2, bkcg = (tid & 3) * 16;     // B loader: 64 rows x 64 halfs
  const bf16* Ag = A + (size_t)(m0 + arow) * K + acg;
  const bf16* Bg = Bt + (size_t)(n0 + bn) * K + bkcg;

  auto stage = [&](int k0, int buf) {
#if defined(HAVE_ASYNC_LDS)
#pragma unroll
    for (int i = 0; i < 4; ++i)
      async_cp16(Ag + k0 + i * 8, &As[buf][arow * 72 + acg + i * 8]);
#pragma unroll
    for (int i = 0; i < 2; ++i)
      async_cp16(Bg + k0 + i * 8, &Bs[buf][bn * 72 + bkcg + i * 8]);
#else
#pragma unroll
    for (int i = 0; i < 4; ++i)
      *(v8bf*)(&As[buf][arow * 72 + acg + i * 8]) = *(const v8bf*)(Ag + k0 + i * 8);
#pragma unroll
    for (int i = 0; i < 2; ++i)
      *(v8bf*)(&Bs[buf][bn * 72 + bkcg + i * 8]) = *(const v8bf*)(Bg + k0 + i * 8);
#endif
  };

  stage(0, 0);
  wait_async_all();
  __syncthreads();

  int nsteps = K >> 6;
  for (int s = 0; s < nsteps; ++s) {
    int buf = s & 1;
    if (s + 1 < nsteps) stage((s + 1) * 64, buf ^ 1);   // prefetch next tile into other buffer

    const bf16* as = &As[buf][0];
    const bf16* bs = &Bs[buf][0];
    int r = lane & 15;
#pragma unroll
    for (int kk = 0; kk < 64; kk += 32) {
      v16bf a0 = ldfragA(as, wm + r, 72, kk, lane);
      v16bf a1 = ldfragA(as, wm + 16 + r, 72, kk, lane);
      v16bf b0 = ldfragB(bs, wn + r, 72, kk, lane);
      v16bf b1 = ldfragB(bs, wn + 16 + r, 72, kk, lane);
      acc[0][0] = wmma_bf16(a0, b0, acc[0][0]);
      acc[0][1] = wmma_bf16(a0, b1, acc[0][1]);
      acc[1][0] = wmma_bf16(a1, b0, acc[1][0]);
      acc[1][1] = wmma_bf16(a1, b1, acc[1][1]);
    }
    wait_async_all();
    __syncthreads();
  }

  int rbase = m0 + wm + ((lane < 16) ? 0 : 8);
#pragma unroll
  for (int ni = 0; ni < 2; ++ni) {
    int col = n0 + wn + ni * 16 + (lane & 15);
    float bvv = bias[col];
#pragma unroll
    for (int mi = 0; mi < 2; ++mi) {
#pragma unroll
      for (int j = 0; j < 8; ++j) {
        int row = rbase + mi * 16 + j;
        float v = acc[mi][ni][j] + bvv;
        if constexpr (EP == 0) {
          ((float*)Cout)[(size_t)row * N + col] = v;
        } else {
          ((bf16*)Cout)[(size_t)row * N + col] = (bf16)gelu_f(v);
        }
      }
    }
  }
}

// --------------------------------------------------------------- RoPE + pack
// qkv f32 [n][3][16][64] -> q/k rotated, per-head bf16 [h][n][64]; V also transposed [h][d][n]
__global__ __launch_bounds__(256) void rope_pack_k(const float* __restrict__ qkv,
                                                   const float* __restrict__ coords,
                                                   const float* __restrict__ invf,
                                                   bf16* __restrict__ qb,
                                                   bf16* __restrict__ kb,
                                                   bf16* __restrict__ vbT) {
  int n = blockIdx.x;
  __shared__ float cs[32], sn[32];
  int t = threadIdx.x;
  if (t < 32) {
    float ang = coords[n * 4 + (t >> 3)] * invf[t];
    cs[t] = cosf(ang);
    sn[t] = sinf(ang);
  }
  __syncthreads();
  const float* q = qkv + (size_t)n * 3 * ED;
  const float* k = q + ED;
  const float* v = k + ED;
  for (int i = t; i < ED; i += 256) {
    int h = i >> 6, d = i & 63;
    size_t o = ((size_t)h * NT + n) * HD + d;
    float qo, ko;
    if (d < 32) {
      qo = q[i] * cs[d] - q[i + 32] * sn[d];
      ko = k[i] * cs[d] - k[i + 32] * sn[d];
    } else {
      int dd = d - 32;
      qo = q[i] * cs[dd] + q[i - 32] * sn[dd];
      ko = k[i] * cs[dd] + k[i - 32] * sn[dd];
    }
    qb[o] = (bf16)qo;
    kb[o] = (bf16)ko;
    vbT[((size_t)h * HD + d) * NT + n] = (bf16)v[i];
  }
}

// -------------------------------------------------------- flash attention
// grid (NT/64, NH); 128 threads (4 waves). Each wave owns 16 query rows.
__global__ __launch_bounds__(128) void attn_k(const bf16* __restrict__ Q,
                                              const bf16* __restrict__ Kk,
                                              const bf16* __restrict__ VT,
                                              const int* __restrict__ seg,
                                              const int* __restrict__ cu,
                                              bf16* __restrict__ Out) {
  __shared__ __align__(16) bf16 qs[64 * 72];    // [m][d]
  __shared__ __align__(16) bf16 ks[64 * 72];    // [key][d]   (B-frag for Q*K^T)
  __shared__ __align__(16) bf16 vsT[64 * 72];   // [d][key]   (B-frag for P*V)
  __shared__ __align__(16) bf16 ps[4][16 * 72]; // per-wave P [row][key]
  __shared__ int segq[64];
  __shared__ int segk[64];

  int h = blockIdx.y;
  int m0 = blockIdx.x * 64;
  int t = threadIdx.x;
  int lane = t & 31, wave = t >> 5;
  int qrow = wave * 16;

  const bf16* Qh = Q + (size_t)h * NT * HD;
  const bf16* Kh = Kk + (size_t)h * NT * HD;
  const bf16* VhT = VT + (size_t)h * HD * NT;   // [d][n]

  {
    int row = t >> 1, cg = (t & 1) * 32;
#pragma unroll
    for (int i = 0; i < 4; ++i)
      *(v8bf*)(qs + row * 72 + cg + i * 8) = *(const v8bf*)(Qh + (size_t)(m0 + row) * HD + cg + i * 8);
    if (t < 64) segq[t] = seg[m0 + t];
  }
  __syncthreads();

  int sf = seg[m0], sl = seg[m0 + 63];
  int kstart = cu[sf] & ~63;
  int kend = cu[sl + 1];
  kend = (kend + 63) & ~63;
  if (kend > NT) kend = NT;

  v8f acc[4];
#pragma unroll
  for (int dt = 0; dt < 4; ++dt) acc[dt] = (v8f){0.f, 0.f, 0.f, 0.f, 0.f, 0.f, 0.f, 0.f};
  float mrow[8], lrow[8];
#pragma unroll
  for (int j = 0; j < 8; ++j) { mrow[j] = -1e30f; lrow[j] = 0.f; }

  for (int kt = kstart; kt < kend; kt += 64) {
    {
      int row = t >> 1, cg = (t & 1) * 32;
#pragma unroll
      for (int i = 0; i < 4; ++i)
        *(v8bf*)(ks + row * 72 + cg + i * 8) =
            *(const v8bf*)(Kh + (size_t)(kt + row) * HD + cg + i * 8);
#pragma unroll
      for (int i = 0; i < 4; ++i)
        *(v8bf*)(vsT + row * 72 + cg + i * 8) =
            *(const v8bf*)(VhT + (size_t)row * NT + kt + cg + i * 8);
      if (t < 64) segk[t] = seg[kt + t];
    }
    __syncthreads();

    // scores S = Q * K^T / 8 (masked by segment equality)
    v8f s[4];
#pragma unroll
    for (int nt = 0; nt < 4; ++nt) {
      v8f sv = {0.f, 0.f, 0.f, 0.f, 0.f, 0.f, 0.f, 0.f};
#pragma unroll
      for (int d0 = 0; d0 < HD; d0 += 32) {
        v16bf a = ldfragA(qs, qrow + (lane & 15), 72, d0, lane);
        v16bf b = ldfragB(ks, nt * 16 + (lane & 15), 72, d0, lane);
        sv = wmma_bf16(a, b, sv);
      }
      int sk = segk[nt * 16 + (lane & 15)];
#pragma unroll
      for (int j = 0; j < 8; ++j) {
        int r = j + ((lane < 16) ? 0 : 8);
        sv[j] = (segq[qrow + r] == sk) ? sv[j] * 0.125f : -1e30f;
      }
      s[nt] = sv;
    }

    // online softmax per row (rows are wave-private; reduce over 16 lanes)
#pragma unroll
    for (int j = 0; j < 8; ++j) {
      float tmax = fmaxf(fmaxf(s[0][j], s[1][j]), fmaxf(s[2][j], s[3][j]));
#pragma unroll
      for (int off = 1; off < 16; off <<= 1) tmax = fmaxf(tmax, __shfl_xor(tmax, off));
      float mnew = fmaxf(mrow[j], tmax);
      float corr = __expf(mrow[j] - mnew);
      float psum = 0.f;
      int r = j + ((lane < 16) ? 0 : 8);
#pragma unroll
      for (int nt = 0; nt < 4; ++nt) {
        float p = __expf(s[nt][j] - mnew);
        psum += p;
        ps[wave][r * 72 + nt * 16 + (lane & 15)] = (bf16)p;
      }
#pragma unroll
      for (int off = 1; off < 16; off <<= 1) psum += __shfl_xor(psum, off);
      lrow[j] = lrow[j] * corr + psum;
      mrow[j] = mnew;
#pragma unroll
      for (int dt = 0; dt < 4; ++dt) acc[dt][j] *= corr;
    }

    // O += P * V
#pragma unroll
    for (int dt = 0; dt < 4; ++dt) {
#pragma unroll
      for (int k0 = 0; k0 < 64; k0 += 32) {
        v16bf a = ldfragA(&ps[wave][0], (lane & 15), 72, k0, lane);
        v16bf b = ldfragB(vsT, dt * 16 + (lane & 15), 72, k0, lane);
        acc[dt] = wmma_bf16(a, b, acc[dt]);
      }
    }
    __syncthreads();
  }

#pragma unroll
  for (int j = 0; j < 8; ++j) {
    float inv = 1.0f / lrow[j];
    int r = j + ((lane < 16) ? 0 : 8);
    int token = m0 + qrow + r;
#pragma unroll
    for (int dt = 0; dt < 4; ++dt)
      Out[(size_t)token * ED + h * HD + dt * 16 + (lane & 15)] = (bf16)(acc[dt][j] * inv);
  }
}

// ----------------------------------------------------- residual + layernorm
__global__ __launch_bounds__(256) void add_ln_k(const float* __restrict__ X,
                                                const float* __restrict__ Y,
                                                const float* __restrict__ g,
                                                const float* __restrict__ b,
                                                float* __restrict__ outf,
                                                bf16* __restrict__ outb) {
  __shared__ float red[256];
  int row = blockIdx.x, t = threadIdx.x;
  float v[4];
  float s = 0.f;
#pragma unroll
  for (int i = 0; i < 4; ++i) {
    int c = t + i * 256;
    v[i] = X[(size_t)row * ED + c] + Y[(size_t)row * ED + c];
    s += v[i];
  }
  red[t] = s;
  __syncthreads();
  for (int o = 128; o > 0; o >>= 1) {
    if (t < o) red[t] += red[t + o];
    __syncthreads();
  }
  float mu = red[0] * (1.0f / ED);
  __syncthreads();
  float s2 = 0.f;
#pragma unroll
  for (int i = 0; i < 4; ++i) {
    v[i] -= mu;
    s2 += v[i] * v[i];
  }
  red[t] = s2;
  __syncthreads();
  for (int o = 128; o > 0; o >>= 1) {
    if (t < o) red[t] += red[t + o];
    __syncthreads();
  }
  float rs = rsqrtf(red[0] * (1.0f / ED) + 1e-5f);
#pragma unroll
  for (int i = 0; i < 4; ++i) {
    int c = t + i * 256;
    float o = v[i] * rs * g[c] + b[c];
    outf[(size_t)row * ED + c] = o;
    if (outb) outb[(size_t)row * ED + c] = (bf16)o;
  }
}

// --------------------------------------------------------------------- host
extern "C" void kernel_launch(void* const* d_in, const int* in_sizes, int n_in,
                              void* d_out, int out_size, void* d_ws, size_t ws_size,
                              hipStream_t stream) {
  const float* coords = (const float*)d_in[0];
  const float* feats  = (const float*)d_in[1];
  const int*   cu     = (const int*)d_in[2];
  const float* Wqkv   = (const float*)d_in[3];
  const float* bqkv   = (const float*)d_in[4];
  const float* Wo     = (const float*)d_in[5];
  const float* bo     = (const float*)d_in[6];
  const float* invf   = (const float*)d_in[7];
  const float* ln1_g  = (const float*)d_in[8];
  const float* ln1_b  = (const float*)d_in[9];
  const float* W1     = (const float*)d_in[10];
  const float* b1     = (const float*)d_in[11];
  const float* W2     = (const float*)d_in[12];
  const float* b2     = (const float*)d_in[13];
  const float* ln2_g  = (const float*)d_in[14];
  const float* ln2_b  = (const float*)d_in[15];
  float* out = (float*)d_out;

  char* ws = (char*)d_ws;
  size_t off = 0;
  auto alloc = [&](size_t bytes) -> void* {
    void* p = ws + off;
    off += (bytes + 255) & ~(size_t)255;
    return p;
  };
  int*  seg    = (int*)alloc(NT * sizeof(int));
  bf16* featsb = (bf16*)alloc((size_t)NT * ED * 2);
  bf16* WqkvT  = (bf16*)alloc((size_t)ED * 3 * ED * 2);    // [3072][1024]
  bf16* WoT    = (bf16*)alloc((size_t)ED * ED * 2);        // [1024][1024]
  bf16* W1T    = (bf16*)alloc((size_t)ED * FFN * 2);       // [4096][1024]
  bf16* W2T    = (bf16*)alloc((size_t)FFN * ED * 2);       // [1024][4096]
  float* qkv   = (float*)alloc((size_t)NT * 3 * ED * 4);
  bf16* qb     = (bf16*)alloc((size_t)NH * NT * HD * 2);
  bf16* kb     = (bf16*)alloc((size_t)NH * NT * HD * 2);
  bf16* vbT    = (bf16*)alloc((size_t)NH * HD * NT * 2);
  bf16* attnob = (bf16*)alloc((size_t)NT * ED * 2);
  float* oproj = (float*)alloc((size_t)NT * ED * 4);
  float* hf    = (float*)alloc((size_t)NT * ED * 4);
  bf16* hb     = (bf16*)alloc((size_t)NT * ED * 2);
  bf16* gelub  = (bf16*)alloc((size_t)NT * FFN * 2);
  float* ffn2  = (float*)alloc((size_t)NT * ED * 4);
  (void)ws_size; (void)in_sizes; (void)n_in; (void)out_size;

  f32_to_bf16_k<<<1024, 256, 0, stream>>>(feats, featsb, NT * ED);
  convT_k<<<dim3(3 * ED / 32, ED / 32), 256, 0, stream>>>(Wqkv, WqkvT, ED, 3 * ED);
  convT_k<<<dim3(ED / 32, ED / 32), 256, 0, stream>>>(Wo, WoT, ED, ED);
  convT_k<<<dim3(FFN / 32, ED / 32), 256, 0, stream>>>(W1, W1T, ED, FFN);
  convT_k<<<dim3(ED / 32, FFN / 32), 256, 0, stream>>>(W2, W2T, FFN, ED);
  seg_k<<<NT / 256, 256, 0, stream>>>(cu, seg);

  // QKV projection: [4096,1024] x [1024,3072]
  gemm_bf16_k<0><<<dim3(3 * ED / 64, NT / 128), 256, 0, stream>>>(featsb, WqkvT, bqkv, qkv, NT, 3 * ED, ED);

  rope_pack_k<<<NT, 256, 0, stream>>>(qkv, coords, invf, qb, kb, vbT);

  attn_k<<<dim3(NT / 64, NH), 128, 0, stream>>>(qb, kb, vbT, seg, cu, attnob);

  // O projection
  gemm_bf16_k<0><<<dim3(ED / 64, NT / 128), 256, 0, stream>>>(attnob, WoT, bo, oproj, NT, ED, ED);

  // h = LN1(feats + oproj), also bf16 copy for FFN
  add_ln_k<<<NT, 256, 0, stream>>>(feats, oproj, ln1_g, ln1_b, hf, hb);

  // FFN1 + GELU -> bf16
  gemm_bf16_k<1><<<dim3(FFN / 64, NT / 128), 256, 0, stream>>>(hb, W1T, b1, gelub, NT, FFN, ED);

  // FFN2
  gemm_bf16_k<0><<<dim3(ED / 64, NT / 128), 256, 0, stream>>>(gelub, W2T, b2, ffn2, NT, ED, FFN);

  // out = LN2(h + ffn2)
  add_ln_k<<<NT, 256, 0, stream>>>(hf, ffn2, ln2_g, ln2_b, out, nullptr);
}
